// DQNAgent_80564996538793
// MI455X (gfx1250) — matrix operand order repaired and checked by reading, not compiled, not run on hardware
//
#include <hip/hip_runtime.h>
#include <hip/hip_bf16.h>

typedef __attribute__((ext_vector_type(16))) __bf16 v16bf;
typedef __attribute__((ext_vector_type(8)))  float  v8f;

#define B_SZ   4096
#define NBLK   100
#define NSLOT  300
#define NYS    1236
#define HID    256
#define STEPS  100
#define NEGV   (-1000.0f)
#define SEQ    1436
#define KPAD   1440
#define M_FEAT (B_SZ * 5) /* 20480 */

static __device__ __forceinline__ float sigm(float x) { return 1.f / (1.f + __expf(-x)); }

// A-fragment for V_WMMA_F32_16X16X32_BF16: lane's 16 halfs are two contiguous
// 16-byte runs at k = hi8 + [0,8) and k = 16 + hi8 + [0,8)  ->  2x ds_load_b128.
__device__ __forceinline__ v16bf load_afrag(const __bf16* rowBase /* + nlo*lda */, int hi8, int kt) {
  union { v16bf v; uint4 q[2]; } av;
  const uint4* p = (const uint4*)(rowBase + kt * 32 + hi8);
  av.q[0] = p[0];   // elements 0..7   (k = hi8 + 0..7)
  av.q[1] = p[2];   // elements 8..15  (k = 16 + hi8 + 0..7)
  return av.v;
}

// ---------------------------------------------------------------------------
// Init kernels
// ---------------------------------------------------------------------------
__global__ void build_state_kernel(const float* __restrict__ to_t, const float* __restrict__ ti_t,
                                   const float* __restrict__ ys_t, float* __restrict__ state_out,
                                   __bf16* __restrict__ Xbf) {
  long i = (long)blockIdx.x * blockDim.x + threadIdx.x;
  if (i >= (long)B_SZ * SEQ * 5) return;
  int r = (int)(i % 5);
  int k = (int)((i / 5) % SEQ);
  int b = (int)(i / (5 * SEQ));
  float v;
  if (k < 100)      v = to_t[((long)b * 100 + k) * 5 + r];
  else if (k < 200) v = ti_t[((long)b * 100 + (k - 100)) * 5 + r];
  else              v = ys_t[((long)b * NYS + (k - 200)) * 5 + r];
  state_out[i] = v;                                   // state [B][1436][5]
  Xbf[((long)b * 5 + r) * KPAD + k] = (__bf16)v;      // X [B*5][1440] bf16 (transposed)
}

__global__ void pad_x_kernel(__bf16* __restrict__ Xbf) {
  int i = blockIdx.x * blockDim.x + threadIdx.x;
  const int npad = KPAD - SEQ; // 4
  if (i >= M_FEAT * npad) return;
  int row = i / npad, p = i % npad;
  Xbf[(long)row * KPAD + SEQ + p] = (__bf16)0.f;
}

// Pre-swizzle W [N][K] (row-major f32) into WMMA B-fragment order bf16:
// dst[((kt*NT + nt)*32 + lane)*16 + e],  n = nt*16 + (lane&15),  k = kt*32 + 8*(lane>>4) + 16*(e>>3) + (e&7)
__global__ void swizzle_w_kernel(const float* __restrict__ W, __bf16* __restrict__ dst,
                                 int N, int K, int NT, long total) {
  long i = (long)blockIdx.x * blockDim.x + threadIdx.x;
  if (i >= total) return;
  int  e    = (int)(i & 15);
  int  lane = (int)((i >> 4) & 31);
  long tile = i >> 9;
  int  nt = (int)(tile % NT);
  int  kt = (int)(tile / NT);
  int  n = nt * 16 + (lane & 15);
  int  k = kt * 32 + ((lane >> 4) << 3) + ((e >> 3) << 4) + (e & 7);
  float v = (n < N && k < K) ? W[(long)n * K + k] : 0.f;
  dst[i] = (__bf16)v;
}

__global__ void slots_init_kernel(const float* __restrict__ slot_info, float* __restrict__ slots3) {
  int i = blockIdx.x * blockDim.x + threadIdx.x;
  if (i >= B_SZ * NSLOT) return;
  slots3[i * 3 + 0] = slot_info[(long)i * 5 + 0];  // length
  slots3[i * 3 + 1] = slot_info[(long)i * 5 + 1];  // width
  slots3[i * 3 + 2] = slot_info[(long)i * 5 + 4];  // count
}

// ---------------------------------------------------------------------------
// Tiled GEMM: relu(A[M,Kpad]bf16 @ Bsw + bias).  Block = 8 waves, 16x128 C tile.
// Whole 16-row A panel staged once into dynamic LDS with uint4 copies, then the
// entire K loop runs barrier-free out of LDS.
// ---------------------------------------------------------------------------
__global__ __launch_bounds__(256) void gemm_bias_relu_kernel(
    const __bf16* __restrict__ A, int lda,       // lda in halfs, multiple of 8
    const __bf16* __restrict__ Bsw, int KT, int NT,
    const float* __restrict__ bias, int N,
    __bf16* __restrict__ ObF, float* __restrict__ Of32) {
  extern __shared__ __align__(16) __bf16 At[];   // 16 * lda halfs
  int t = threadIdx.x, lane = t & 31, wave = t >> 5;
  int nlo = lane & 15, hi8 = (lane >> 4) << 3;
  long mbase = (long)blockIdx.x * 16;
  int  nt = blockIdx.y * 8 + wave;
  int  n  = nt * 16 + nlo;

  // --- stage A panel: 16 rows x lda halfs, 16B chunks, fully coalesced
  int chunksPerRow = lda >> 3;                   // uint4 = 8 halfs
  int totalChunks = 16 * chunksPerRow;
  const uint4* Ag = (const uint4*)A;
  uint4* AtV = (uint4*)At;
  for (int c = t; c < totalChunks; c += 256) {
    int m = c / chunksPerRow, pos = c - m * chunksPerRow;
    AtV[c] = Ag[(mbase + m) * chunksPerRow + pos];
  }
  __syncthreads();

  const __bf16* rowBase = At + nlo * lda;
  v8f acc = {};
#pragma unroll 4
  for (int kt = 0; kt < KT; ++kt) {
    v16bf av = load_afrag(rowBase, hi8, kt);
    v16bf bv = *(const v16bf*)(Bsw + (((long)kt * NT + nt) * 32 + lane) * 16);
    acc = __builtin_amdgcn_wmma_f32_16x16x32_bf16(false, av, false, bv, (short)0, acc, false, false);
  }
  float bval = (n < N) ? bias[n] : 0.f;
#pragma unroll
  for (int v = 0; v < 8; ++v) {                  // C layout: m = v + 8*(lane>>4), n = lane&15
    float x = fmaxf(acc[v] + bval, 0.f);
    long m = mbase + v + hi8;
    if (n < N) {
      if (ObF)  ObF[m * N + n]  = (__bf16)x;
      if (Of32) Of32[m * N + n] = x;
    }
  }
}

__global__ void feat_kernel(const float* __restrict__ H3, const float* __restrict__ Wfe,
                            const float* __restrict__ bfe, float* __restrict__ feat) {
  int i = blockIdx.x * blockDim.x + threadIdx.x;
  if (i >= B_SZ * HID) return;
  int b = i >> 8, j = i & 255;
  float s = bfe[0];
#pragma unroll
  for (int r = 0; r < 5; ++r) s += H3[((long)b * 5 + r) * HID + j] * Wfe[r];
  feat[i] = s;
}

// ---------------------------------------------------------------------------
// One LSTM cell step for 16 batch rows. Wave w owns gate columns j in [32w,32w+32):
// ntiles {g*16 + 2w, g*16 + 2w + 1} for gates g=0..3 -> i/f/g/o stay in registers.
// ---------------------------------------------------------------------------
__device__ __forceinline__ void lstm_step(
    __bf16* h, float* c_reg,
    const __bf16* __restrict__ Whsw,
    const float* __restrict__ Wih, const float* __restrict__ bih, const float* __restrict__ bhh,
    const float* inp, int wave, int lane) {
  int nlo = lane & 15, hi8 = (lane >> 4) << 3;
  v8f acc[4][2];
#pragma unroll
  for (int g = 0; g < 4; ++g)
#pragma unroll
    for (int u2 = 0; u2 < 2; ++u2) {
      int u = 2 * wave + u2;
      int n = g * 256 + u * 16 + nlo;
      float bval = bih[n] + bhh[n];
      v8f a;
      if (inp) {                       // fold x@Wih^T (K=5) into accumulator seed
        float w0 = Wih[n * 5 + 0], w1 = Wih[n * 5 + 1], w2 = Wih[n * 5 + 2];
        float w3 = Wih[n * 5 + 3], w4 = Wih[n * 5 + 4];
#pragma unroll
        for (int v = 0; v < 8; ++v) {
          int m = v + hi8;
          a[v] = bval + inp[m * 8 + 0] * w0 + inp[m * 8 + 1] * w1 + inp[m * 8 + 2] * w2 +
                 inp[m * 8 + 3] * w3 + inp[m * 8 + 4] * w4;
        }
      } else {
#pragma unroll
        for (int v = 0; v < 8; ++v) a[v] = bval;
      }
      acc[g][u2] = a;
    }
  const __bf16* rowBase = h + nlo * HID;
#pragma unroll
  for (int kt = 0; kt < 8; ++kt) {
    v16bf av = load_afrag(rowBase, hi8, kt);
#pragma unroll
    for (int g = 0; g < 4; ++g)
#pragma unroll
      for (int u2 = 0; u2 < 2; ++u2) {
        int nt = g * 16 + 2 * wave + u2;
        v16bf bv = *(const v16bf*)(Whsw + (((long)kt * 64 + nt) * 32 + lane) * 16);
        acc[g][u2] = __builtin_amdgcn_wmma_f32_16x16x32_bf16(false, av, false, bv, (short)0,
                                                             acc[g][u2], false, false);
      }
  }
  __syncthreads();  // all waves done reading h before overwrite
#pragma unroll
  for (int u2 = 0; u2 < 2; ++u2) {
    int j = (2 * wave + u2) * 16 + nlo;
#pragma unroll
    for (int v = 0; v < 8; ++v) {
      int m = v + hi8;
      float ii = sigm(acc[0][u2][v]);
      float ff = sigm(acc[1][u2][v]);
      float gg = tanhf(acc[2][u2][v]);
      float oo = sigm(acc[3][u2][v]);
      float c = ff * c_reg[u2 * 8 + v] + ii * gg;
      c_reg[u2 * 8 + v] = c;
      h[m * HID + j] = (__bf16)(oo * tanhf(c));
    }
  }
}

__device__ __forceinline__ void logits_mm(
    const __bf16* h, const __bf16* __restrict__ Bsw, int NT,
    const float* __restrict__ bias, int N, float* L, int wave, int lane) {
  int nlo = lane & 15, hi8 = (lane >> 4) << 3;
  const __bf16* rowBase = h + nlo * HID;
  for (int nt = wave; nt < NT; nt += 8) {
    int n = nt * 16 + nlo;
    float bval = (n < N) ? bias[n] : 0.f;
    v8f acc;
#pragma unroll
    for (int v = 0; v < 8; ++v) acc[v] = bval;
#pragma unroll
    for (int kt = 0; kt < 8; ++kt) {
      v16bf av = load_afrag(rowBase, hi8, kt);
      v16bf bv = *(const v16bf*)(Bsw + (((long)kt * NT + nt) * 32 + lane) * 16);
      acc = __builtin_amdgcn_wmma_f32_16x16x32_bf16(false, av, false, bv, (short)0, acc, false, false);
    }
#pragma unroll
    for (int v = 0; v < 8; ++v) L[(v + hi8) * 304 + n] = acc[v];
  }
}

// ---------------------------------------------------------------------------
// Persistent recurrent kernel: 16 batch rows per WG, 100 steps
// ---------------------------------------------------------------------------
__global__ __launch_bounds__(256) void recurrent_kernel(
    const float* __restrict__ feat,
    const float* __restrict__ to_t, const float* __restrict__ ti_t,
    const __bf16* __restrict__ WhToSw, const __bf16* __restrict__ WhTiSw,
    const __bf16* __restrict__ WbSw, const __bf16* __restrict__ WsSw,
    const float* __restrict__ Wih_to, const float* __restrict__ bih_to, const float* __restrict__ bhh_to,
    const float* __restrict__ bih_ti, const float* __restrict__ bhh_ti,
    const float* __restrict__ bb, const float* __restrict__ bs,
    float* __restrict__ slots3, float* __restrict__ actions) {
  __shared__ __align__(16) __bf16 h_to[16 * HID];
  __shared__ __align__(16) __bf16 h_ti[16 * HID];
  __shared__ float  L[16 * 304];
  __shared__ float  inp[16 * 8];
  __shared__ unsigned char m_to[16 * 112];
  __shared__ unsigned char m_ti[16 * 112];
  __shared__ int selb[16], sels[16];

  int t = threadIdx.x, lane = t & 31, wave = t >> 5;
  long b0 = (long)blockIdx.x * 16;

  for (int idx = t; idx < 16 * HID; idx += 256) {
    int r = idx >> 8, j = idx & 255;
    float f = feat[(b0 + r) * HID + j];
    h_to[idx] = (__bf16)f;
    h_ti[idx] = (__bf16)f;
  }
  for (int idx = t; idx < 16 * 112; idx += 256) {
    int r = idx / 112, j = idx % 112;
    unsigned char a = 0, c = 0;
    if (j < NBLK) {
      a = (to_t[((b0 + r) * NBLK + j) * 5] != -1.0f) ? 1 : 0;
      c = (ti_t[((b0 + r) * NBLK + j) * 5] != -1.0f) ? 1 : 0;
    }
    m_to[idx] = a; m_ti[idx] = c;
  }
  for (int idx = t; idx < 16 * 8; idx += 256) inp[idx] = 0.f;
  float c_to[16], c_ti[16];
#pragma unroll
  for (int q = 0; q < 16; ++q) { c_to[q] = 0.f; c_ti[q] = 0.f; }
  __syncthreads();

  for (int step = 0; step < STEPS; ++step) {
    // ---- LSTM(to)
    lstm_step(h_to, c_to, WhToSw, Wih_to, bih_to, bhh_to, inp, wave, lane);
    __syncthreads();
    // ---- block logits (to) + argmax
    logits_mm(h_to, WbSw, 7, bb, NBLK, L, wave, lane);
    __syncthreads();
    if (t < 16) {
      float best = -3.4e38f; int bi = 0;
      for (int n2 = 0; n2 < NBLK; ++n2) {
        float v = m_to[t * 112 + n2] ? L[t * 304 + n2] : NEGV;
        if (v > best) { best = v; bi = n2; }
      }
      selb[t] = bi; m_to[t * 112 + bi] = 0;
      for (int r5 = 0; r5 < 5; ++r5)
        inp[t * 8 + r5] = to_t[((b0 + t) * NBLK + bi) * 5 + r5];
    }
    __syncthreads();
    // ---- slot logits + feasibility mask + argmax + slot update
    logits_mm(h_to, WsSw, 19, bs, NSLOT, L, wave, lane);
    __syncthreads();
    if (t < 16) {
      long b = b0 + t;
      float l0 = inp[t * 8 + 0], w0 = inp[t * 8 + 1];
      float best = -3.4e38f; int bi = 0;
      for (int s = 0; s < NSLOT; ++s) {
        const float* sp = slots3 + (b * NSLOT + s) * 3;
        float sl = sp[0], sw = sp[1];
        float v = ((l0 >= sl) || (w0 >= sw) || (sl == 0.f)) ? NEGV : L[t * 304 + s];
        if (v > best) { best = v; bi = s; }
      }
      sels[t] = bi;
      float* sp = slots3 + (b * NSLOT + bi) * 3;
      float cnt = sp[2] - 1.f;
      if (cnt == 0.f) { sp[0] = 0.f; sp[1] = 0.f; sp[2] = 0.f; }
      else sp[2] = cnt;
    }
    __syncthreads();
    // ---- LSTM(ti), zero input
    lstm_step(h_ti, c_ti, WhTiSw, nullptr, bih_ti, bhh_ti, nullptr, wave, lane);
    __syncthreads();
    logits_mm(h_ti, WbSw, 7, bb, NBLK, L, wave, lane);
    __syncthreads();
    if (t < 16) {
      float best = -3.4e38f; int bi = 0;
      for (int n2 = 0; n2 < NBLK; ++n2) {
        float v = m_ti[t * 112 + n2] ? L[t * 304 + n2] : NEGV;
        if (v > best) { best = v; bi = n2; }
      }
      m_ti[t * 112 + bi] = 0;
      float* o = actions + ((b0 + t) * STEPS + step) * 3;
      o[0] = (float)selb[t]; o[1] = (float)sels[t]; o[2] = (float)bi;
    }
    __syncthreads();
  }
}

// ---------------------------------------------------------------------------
// Launcher
// ---------------------------------------------------------------------------
extern "C" void kernel_launch(void* const* d_in, const int* in_sizes, int n_in,
                              void* d_out, int out_size, void* d_ws, size_t ws_size,
                              hipStream_t stream) {
  const float* to_t = (const float*)d_in[0];
  const float* ti_t = (const float*)d_in[1];
  const float* ys_t = (const float*)d_in[2];
  const float* slot_info = (const float*)d_in[3];
  const float* W1 = (const float*)d_in[4];   const float* b1 = (const float*)d_in[5];
  const float* W2 = (const float*)d_in[6];   const float* b2 = (const float*)d_in[7];
  const float* W3 = (const float*)d_in[8];   const float* b3 = (const float*)d_in[9];
  const float* Wfe2 = (const float*)d_in[10]; const float* bfe2 = (const float*)d_in[11];
  const float* Wih_to = (const float*)d_in[12]; const float* Whh_to = (const float*)d_in[13];
  const float* bih_to = (const float*)d_in[14]; const float* bhh_to = (const float*)d_in[15];
  const float* Wih_ti = (const float*)d_in[16]; const float* Whh_ti = (const float*)d_in[17];
  const float* bih_ti = (const float*)d_in[18]; const float* bhh_ti = (const float*)d_in[19];
  const float* Wb = (const float*)d_in[20]; const float* bb = (const float*)d_in[21];
  const float* Ws = (const float*)d_in[22]; const float* bs = (const float*)d_in[23];
  (void)Wih_ti; (void)in_sizes; (void)n_in; (void)out_size; (void)ws_size;

  float* outf = (float*)d_out;
  float* actions = outf + (long)B_SZ * SEQ * 5;

  char* base = (char*)d_ws;
  size_t off = 0;
  auto alloc = [&](size_t bytes) -> void* {
    void* p = base + off;
    off = (off + bytes + 255) & ~(size_t)255;
    return p;
  };
  __bf16* Xbf    = (__bf16*)alloc((size_t)M_FEAT * KPAD * 2);
  __bf16* H1bf   = (__bf16*)alloc((size_t)M_FEAT * 512 * 2);
  __bf16* H2bf   = (__bf16*)alloc((size_t)M_FEAT * 512 * 2);
  float*  H3f    = (float*) alloc((size_t)M_FEAT * 256 * 4);
  float*  feat   = (float*) alloc((size_t)B_SZ * HID * 4);
  __bf16* W1sw   = (__bf16*)alloc((size_t)45 * 32 * 512 * 2);
  __bf16* W2sw   = (__bf16*)alloc((size_t)16 * 32 * 512 * 2);
  __bf16* W3sw   = (__bf16*)alloc((size_t)16 * 16 * 512 * 2);
  __bf16* WhToSw = (__bf16*)alloc((size_t)8 * 64 * 512 * 2);
  __bf16* WhTiSw = (__bf16*)alloc((size_t)8 * 64 * 512 * 2);
  __bf16* WbSw   = (__bf16*)alloc((size_t)8 * 7 * 512 * 2);
  __bf16* WsSw   = (__bf16*)alloc((size_t)8 * 19 * 512 * 2);
  float*  slots3 = (float*) alloc((size_t)B_SZ * NSLOT * 3 * 4);

  long nstate = (long)B_SZ * SEQ * 5;
  build_state_kernel<<<(unsigned)((nstate + 255) / 256), 256, 0, stream>>>(to_t, ti_t, ys_t, outf, Xbf);
  pad_x_kernel<<<(M_FEAT * 4 + 255) / 256, 256, 0, stream>>>(Xbf);

  auto swz = [&](const float* W, __bf16* dst, int N, int K, int NT, int KT) {
    long total = (long)KT * NT * 512;
    swizzle_w_kernel<<<(unsigned)((total + 255) / 256), 256, 0, stream>>>(W, dst, N, K, NT, total);
  };
  swz(W1, W1sw, 512, 1436, 32, 45);
  swz(W2, W2sw, 512, 512, 32, 16);
  swz(W3, W3sw, 256, 512, 16, 16);
  swz(Whh_to, WhToSw, 1024, 256, 64, 8);
  swz(Whh_ti, WhTiSw, 1024, 256, 64, 8);
  swz(Wb, WbSw, 100, 256, 7, 8);
  swz(Ws, WsSw, 300, 256, 19, 8);
  slots_init_kernel<<<(B_SZ * NSLOT + 255) / 256, 256, 0, stream>>>(slot_info, slots3);

  gemm_bias_relu_kernel<<<dim3(M_FEAT / 16, 4), 256, (size_t)16 * KPAD * 2, stream>>>(
      Xbf, KPAD, W1sw, 45, 32, b1, 512, H1bf, nullptr);
  gemm_bias_relu_kernel<<<dim3(M_FEAT / 16, 4), 256, (size_t)16 * 512 * 2, stream>>>(
      H1bf, 512, W2sw, 16, 32, b2, 512, H2bf, nullptr);
  gemm_bias_relu_kernel<<<dim3(M_FEAT / 16, 2), 256, (size_t)16 * 512 * 2, stream>>>(
      H2bf, 512, W3sw, 16, 16, b3, 256, nullptr, H3f);
  feat_kernel<<<(B_SZ * HID + 255) / 256, 256, 0, stream>>>(H3f, Wfe2, bfe2, feat);

  recurrent_kernel<<<B_SZ / 16, 256, 0, stream>>>(feat, to_t, ti_t, WhToSw, WhTiSw, WbSw, WsSw,
      Wih_to, bih_to, bhh_to, bih_ti, bhh_ti, bb, bs, slots3, actions);
}
